// MultiModelFusionSystem_80745385165510
// MI455X (gfx1250) — matrix operand order repaired and checked by reading, not compiled, not run on hardware
//
#include <hip/hip_runtime.h>
#include <hip/hip_bf16.h>
#include <stdint.h>

#define E_  4
#define N_  2048
#define D_  512
#define H_  8
#define DH_ 64
#define FF_ 2048
#define L_  4
#define M_  4096

typedef __attribute__((ext_vector_type(16))) __bf16       v16bf;
typedef __attribute__((ext_vector_type(8)))  float        v8f;
typedef __attribute__((ext_vector_type(4)))  unsigned int v4u;
typedef unsigned short u16;

union FragBF { v4u u[2]; v16bf v; };

__device__ __forceinline__ u16 f2bf(float f) {
  unsigned u = __float_as_uint(f);
  u += 0x7FFFu + ((u >> 16) & 1u);          // round-to-nearest-even
  return (u16)(u >> 16);
}

#define EP_RELU 1
#define EP_CLIP 2

// C[M,N] = epilogue( alpha * A[M,K](bf16) @ B[N,K](bf16)^T )
// Block tile 128(M) x 64(N): 4 waves in 2x2; wave tile 64x32 = 4x2 WMMA frags.
// A-frag layout (16x32 bf16): lane m=lane&15, elems 0..7 = K+kh*8.., 8..15 = K+16+kh*8..
// B-frag layout (32x16 bf16): lane n=lane&15, elems 0..7 = K+kh*8.., 8..15 = K+16+kh*8..
// C layout: vgpr r, lane l -> row r+(l>>4)*8, col l&15
__global__ __launch_bounds__(128)
void mmfs_gemm_nt(const u16* __restrict__ A, int lda,
                  const u16* __restrict__ B, int ldb,
                  int Kdim, float alpha, int ep, float clipv,
                  const float* __restrict__ bias,
                  const float* __restrict__ addend, int ldadd,
                  float* __restrict__ Cf, int ldcf,
                  u16* __restrict__ Cb, int ldcb)
{
  const int wave = threadIdx.x >> 5;
  const int lane = threadIdx.x & 31;
  const int m0 = blockIdx.x * 128 + (wave >> 1) * 64;
  const int n0 = blockIdx.y * 64  + (wave & 1) * 32;
  const int fr = lane & 15;
  const int kh = lane >> 4;

  v8f acc[4][2];
#pragma unroll
  for (int i = 0; i < 4; i++)
#pragma unroll
    for (int j = 0; j < 2; j++)
#pragma unroll
      for (int r = 0; r < 8; r++) acc[i][j][r] = 0.f;

  const u16* aR[4];
  const u16* bR[2];
#pragma unroll
  for (int i = 0; i < 4; i++) aR[i] = A + (size_t)(m0 + 16 * i + fr) * lda;
#pragma unroll
  for (int j = 0; j < 2; j++) bR[j] = B + (size_t)(n0 + 16 * j + fr) * ldb;

  for (int k = 0; k < Kdim; k += 32) {
    FragBF a[4], b[2];
    const int ka = k + kh * 8;
    const int kb = k + kh * 8;
#pragma unroll
    for (int j = 0; j < 2; j++) {
      b[j].u[0] = *(const v4u*)(bR[j] + kb);
      b[j].u[1] = *(const v4u*)(bR[j] + kb + 16);
    }
#pragma unroll
    for (int i = 0; i < 4; i++) {
      a[i].u[0] = *(const v4u*)(aR[i] + ka);
      a[i].u[1] = *(const v4u*)(aR[i] + ka + 16);
    }
    if (k + 256 < Kdim) {            // uniform: stream-prefetch ~8 iters ahead
#pragma unroll
      for (int i = 0; i < 4; i++) __builtin_prefetch(aR[i] + ka + 256, 0, 3);
#pragma unroll
      for (int j = 0; j < 2; j++) __builtin_prefetch(bR[j] + kb + 256, 0, 3);
    }
#pragma unroll
    for (int i = 0; i < 4; i++) {
      acc[i][0] = __builtin_amdgcn_wmma_f32_16x16x32_bf16(false, a[i].v, false, b[0].v, (short)0, acc[i][0], false, false);
      acc[i][1] = __builtin_amdgcn_wmma_f32_16x16x32_bf16(false, a[i].v, false, b[1].v, (short)0, acc[i][1], false, false);
    }
  }

  const int colL = lane & 15;
  const int rowB = (lane >> 4) * 8;
#pragma unroll
  for (int i = 0; i < 4; i++)
#pragma unroll
    for (int j = 0; j < 2; j++) {
      const int col = n0 + j * 16 + colL;
      const float bv = bias ? bias[col] : 0.f;
#pragma unroll
      for (int r = 0; r < 8; r++) {
        const int row = m0 + i * 16 + rowB + r;
        float v = acc[i][j][r] * alpha + bv;
        if (ep & EP_RELU) v = fmaxf(v, 0.f);
        if (ep & EP_CLIP) v = fminf(fmaxf(v, -clipv), clipv);
        if (addend) v += addend[(size_t)row * ldadd + col];
        if (Cf) Cf[(size_t)row * ldcf + col] = v;
        if (Cb) Cb[(size_t)row * ldcb + col] = f2bf(v);
      }
    }
}

// Row softmax (optional mask range excluded), f32 in -> bf16 out. C <= 8192.
__global__ __launch_bounds__(256)
void mmfs_softmax(const float* __restrict__ src, int ldsrc, int C,
                  int maskStart, int maskLen,
                  u16* __restrict__ dst, int lddst)
{
  __shared__ float red[256];
  __shared__ float cache[8192];
  const size_t row = blockIdx.x;
  const float* p = src + row * (size_t)ldsrc;
  const int t = threadIdx.x;

  float mx = -3.0e38f;
  for (int c = t; c < C; c += 256) {
    float v = p[c];
    if (c >= maskStart && c < maskStart + maskLen) v = -3.0e38f;
    cache[c] = v;
    mx = fmaxf(mx, v);
  }
  red[t] = mx; __syncthreads();
  for (int s = 128; s > 0; s >>= 1) { if (t < s) red[t] = fmaxf(red[t], red[t + s]); __syncthreads(); }
  mx = red[0]; __syncthreads();

  float sum = 0.f;
  for (int c = t; c < C; c += 256) {
    const float v = cache[c];
    const float e = (v <= -1.0e37f) ? 0.f : __expf(v - mx);
    cache[c] = e;
    sum += e;
  }
  red[t] = sum; __syncthreads();
  for (int s = 128; s > 0; s >>= 1) { if (t < s) red[t] += red[t + s]; __syncthreads(); }
  const float inv = 1.f / red[0];
  __syncthreads();
  for (int c = t; c < C; c += 256)
    dst[row * (size_t)lddst + c] = f2bf(cache[c] * inv);
}

// LayerNorm over 512 cols, optional affine, f32 + bf16 outputs.
__global__ __launch_bounds__(256)
void mmfs_ln512(const float* __restrict__ src, int ldsrc,
                const float* __restrict__ gamma, const float* __restrict__ beta,
                float* __restrict__ dstF, int ldf,
                u16* __restrict__ dstB, int ldb2)
{
  __shared__ float red[256];
  const size_t row = blockIdx.x;
  const float* p = src + row * (size_t)ldsrc;
  const int t = threadIdx.x;
  const float v0 = p[t], v1 = p[t + 256];
  red[t] = v0 + v1; __syncthreads();
  for (int s = 128; s > 0; s >>= 1) { if (t < s) red[t] += red[t + s]; __syncthreads(); }
  const float mu = red[0] * (1.f / 512.f); __syncthreads();
  const float d0 = v0 - mu, d1 = v1 - mu;
  red[t] = d0 * d0 + d1 * d1; __syncthreads();
  for (int s = 128; s > 0; s >>= 1) { if (t < s) red[t] += red[t + s]; __syncthreads(); }
  const float rstd = rsqrtf(red[0] * (1.f / 512.f) + 1e-5f);
  float o0 = d0 * rstd, o1 = d1 * rstd;
  if (gamma) { o0 = o0 * gamma[t] + beta[t]; o1 = o1 * gamma[t + 256] + beta[t + 256]; }
  if (dstF) { dstF[row * (size_t)ldf + t] = o0; dstF[row * (size_t)ldf + t + 256] = o1; }
  if (dstB) { dstB[row * (size_t)ldb2 + t] = f2bf(o0); dstB[row * (size_t)ldb2 + t + 256] = f2bf(o1); }
}

// 32x32 LDS tile transpose, bf16, arbitrary leading dims. grid=(R/32, C/32)
__global__ __launch_bounds__(256)
void mmfs_transpose(const u16* __restrict__ src, int ldsrc,
                    u16* __restrict__ dst, int lddst)
{
  __shared__ u16 tile[32][33];
  const int r0 = blockIdx.x * 32, c0 = blockIdx.y * 32;
  const int tx = threadIdx.x & 31, ty = threadIdx.x >> 5;
  for (int i = ty; i < 32; i += 8)
    tile[i][tx] = src[(size_t)(r0 + i) * ldsrc + c0 + tx];
  __syncthreads();
  for (int i = ty; i < 32; i += 8)
    dst[(size_t)(c0 + i) * lddst + r0 + tx] = tile[tx][i];
}

__global__ void mmfs_cvt(const float* __restrict__ s, u16* __restrict__ d, size_t n)
{
  size_t i = (size_t)blockIdx.x * 256 + threadIdx.x;
  const size_t st = (size_t)gridDim.x * 256;
  for (; i < n; i += st) d[i] = f2bf(s[i]);
}

__global__ void mmfs_pool(const float* __restrict__ outs, float* __restrict__ pooled)
{
  const int idx = blockIdx.x * 256 + threadIdx.x;       // 0..E*D-1
  const int e = idx >> 9, d = idx & 511;
  const float* p = outs + (size_t)e * N_ * D_ + d;
  float s = 0.f;
  for (int n = 0; n < N_; n++) s += p[(size_t)n * D_];
  pooled[idx] = s * (1.f / (float)N_);
}

__global__ __launch_bounds__(128)
void mmfs_gate(const float* __restrict__ pooled, const float* __restrict__ gate,
               float* __restrict__ gw_out)
{
  __shared__ float red[128];
  const int t = threadIdx.x;
  const int e = t >> 5, l = t & 31;
  float s = 0.f;
  for (int d = l; d < D_; d += 32) s += pooled[e * D_ + d] * gate[d];
  red[t] = s; __syncthreads();
  for (int sh = 16; sh > 0; sh >>= 1) { if (l < sh) red[t] += red[t + sh]; __syncthreads(); }
  if (t == 0) {
    float lg[E_], m = -3.0e38f;
    for (int i = 0; i < E_; i++) { lg[i] = fminf(fmaxf(red[i * 32], -10.f), 10.f); m = fmaxf(m, lg[i]); }
    float sm = 0.f;
    for (int i = 0; i < E_; i++) { lg[i] = __expf(lg[i] - m); sm += lg[i]; }
    for (int i = 0; i < E_; i++) gw_out[i] = lg[i] / sm;
  }
}

__global__ void mmfs_fuse(const float* __restrict__ outs, const float* __restrict__ gw,
                          float* __restrict__ fused)
{
  const size_t i = (size_t)blockIdx.x * 256 + threadIdx.x;
  const size_t tot = (size_t)N_ * D_;
  if (i >= tot) return;
  float s = 0.f;
  for (int e = 0; e < E_; e++) s += gw[e] * outs[(size_t)e * tot + i];
  fused[i] = s;
}

__global__ void mmfs_ema(const float* __restrict__ mem, const float* __restrict__ upd,
                         float* __restrict__ outm, size_t n)
{
  size_t i = (size_t)blockIdx.x * 256 + threadIdx.x;
  const size_t st = (size_t)gridDim.x * 256;
  for (; i < n; i += st) {
    const float u = fminf(fmaxf(0.1f * upd[i], -0.1f), 0.1f);
    outm[i] = 0.9f * mem[i] + u;
  }
}

extern "C" void kernel_launch(void* const* d_in, const int* in_sizes, int n_in,
                              void* d_out, int out_size, void* d_ws, size_t ws_size,
                              hipStream_t stream)
{
  (void)in_sizes; (void)n_in; (void)out_size; (void)ws_size;
  const float* tokens  = (const float*)d_in[0];
  const float* memsrc  = (const float*)d_in[1];
  const float* ipw     = (const float*)d_in[2];
  const float* ipb     = (const float*)d_in[3];
  const float* opw     = (const float*)d_in[4];
  const float* opb     = (const float*)d_in[5];
  const float* w1      = (const float*)d_in[6];
  const float* b1      = (const float*)d_in[7];
  const float* w2      = (const float*)d_in[8];
  const float* b2      = (const float*)d_in[9];
  const float* n1w     = (const float*)d_in[10];
  const float* n1b     = (const float*)d_in[11];
  const float* n2w     = (const float*)d_in[12];
  const float* n2b     = (const float*)d_in[13];
  const float* aggw    = (const float*)d_in[14];
  const float* aggb    = (const float*)d_in[15];
  const float* ggate   = (const float*)d_in[16];

  float* out_fused = (float*)d_out;                       // [N, D]
  float* out_gw    = out_fused + (size_t)N_ * D_;         // [E]
  float* out_mem   = out_gw + E_;                         // [L, M, D]

  // ---- workspace bump allocator (256B aligned) ----
  char* wsb = (char*)d_ws;
  size_t off = 0;
  auto alloc = [&](size_t bytes) -> void* {
    off = (off + 255) & ~(size_t)255;
    void* p = wsb + off;
    off += bytes;
    return p;
  };
  const size_t ND = (size_t)N_ * D_;

  u16* ipw_b  = (u16*)alloc((size_t)E_ * 3 * D_ * D_ * 2);
  u16* opw_b  = (u16*)alloc((size_t)E_ * D_ * D_ * 2);
  u16* w1_b   = (u16*)alloc((size_t)E_ * FF_ * D_ * 2);
  u16* w2_b   = (u16*)alloc((size_t)E_ * D_ * FF_ * 2);
  u16* aggw_b = (u16*)alloc((size_t)D_ * L_ * D_ * 2);
  u16* mem_b  = (u16*)alloc((size_t)L_ * M_ * D_ * 2);
  u16* memT_b = (u16*)alloc((size_t)L_ * D_ * M_ * 2);

  float* xln_f  = (float*)alloc((size_t)E_ * ND * 4);
  u16*   xln_b  = (u16*)  alloc((size_t)E_ * ND * 2);
  u16*   xlnT_b = (u16*)  alloc((size_t)D_ * E_ * N_ * 2);

  // per-expert temporaries (reused across the e-loop)
  float* x1f   = (float*)alloc(ND * 4);
  u16*   x1b   = (u16*)  alloc(ND * 2);
  u16*   qkv_b = (u16*)  alloc((size_t)N_ * 3 * D_ * 2);
  u16*   vt_b  = (u16*)  alloc((size_t)DH_ * N_ * 2);
  u16*   sa_b  = (u16*)  alloc(ND * 2);
  float* t0    = (float*)alloc(ND * 4);
  float* x2f   = (float*)alloc(ND * 4);
  u16*   x2b   = (u16*)  alloc(ND * 2);
  u16*   h1_b  = (u16*)  alloc((size_t)N_ * FF_ * 2);
  float* x3f   = (float*)alloc(ND * 4);
  u16*   x3b   = (u16*)  alloc(ND * 2);
  u16*   x3Tb  = (u16*)  alloc((size_t)D_ * N_ * 2);
  u16*   retr_b= (u16*)  alloc((size_t)N_ * L_ * D_ * 2);

  float* S1    = (float*)alloc((size_t)N_ * 8192 * 4);    // logits / scores / ml
  u16*   S2    = (u16*)  alloc((size_t)N_ * 8192 * 2);    // probs (bf16)
  u16*   pT_b  = (u16*)  alloc((size_t)M_ * N_ * 2);      // probs^T for upd
  float* updsum= (float*)alloc((size_t)L_ * M_ * D_ * 4);
  float* outs  = (float*)alloc((size_t)E_ * ND * 4);
  float* pooled= (float*)alloc((size_t)E_ * D_ * 4);

  const float invSqrtD = 0.044194173824159216f;  // 1/sqrt(512)

  auto cvt = [&](const float* s, u16* d, size_t n) {
    int g = (int)((n + 255) / 256); if (g > 8192) g = 8192;
    mmfs_cvt<<<g, 256, 0, stream>>>(s, d, n);
  };
  auto gemm = [&](const u16* A, int lda, const u16* B, int ldb,
                  int Md, int Nd, int Kd, float alpha, int ep, float clipv,
                  const float* bias, const float* addend, int ldadd,
                  float* Cf, int ldcf, u16* Cb, int ldcb) {
    dim3 g(Md / 128, Nd / 64), b(128);
    mmfs_gemm_nt<<<g, b, 0, stream>>>(A, lda, B, ldb, Kd, alpha, ep, clipv,
                                      bias, addend, ldadd, Cf, ldcf, Cb, ldcb);
  };
  auto tr = [&](const u16* s, int lds_, u16* d, int ldd_, int R, int C) {
    dim3 g(R / 32, C / 32);
    mmfs_transpose<<<g, 256, 0, stream>>>(s, lds_, d, ldd_);
  };

  // ---- stage A: weight conversion, token LN, transposes ----
  cvt(ipw,    ipw_b,  (size_t)E_ * 3 * D_ * D_);
  cvt(opw,    opw_b,  (size_t)E_ * D_ * D_);
  cvt(w1,     w1_b,   (size_t)E_ * FF_ * D_);
  cvt(w2,     w2_b,   (size_t)E_ * D_ * FF_);
  cvt(aggw,   aggw_b, (size_t)D_ * L_ * D_);
  cvt(memsrc, mem_b,  (size_t)L_ * M_ * D_);
  for (int l = 0; l < L_; l++)
    tr(mem_b + (size_t)l * M_ * D_, D_, memT_b + (size_t)l * D_ * M_, M_, M_, D_);

  mmfs_ln512<<<E_ * N_, 256, 0, stream>>>(tokens, D_, nullptr, nullptr,
                                          xln_f, D_, xln_b, D_);
  tr(xln_b, D_, xlnT_b, E_ * N_, E_ * N_, D_);
  hipMemsetAsync(updsum, 0, (size_t)L_ * M_ * D_ * 4, stream);

  // ---- per-expert pipeline ----
  for (int e = 0; e < E_; e++) {
    const u16*   xe_b = xln_b + (size_t)e * ND;
    const float* xe_f = xln_f + (size_t)e * ND;

    // cross-expert attention (full 4N keys, self block masked == softmax over 3N ctx)
    gemm(xe_b, D_, xln_b, D_, N_, E_ * N_, D_, invSqrtD, EP_CLIP, 10.f,
         nullptr, nullptr, 0, S1, E_ * N_, nullptr, 0);
    mmfs_softmax<<<N_, 256, 0, stream>>>(S1, E_ * N_, E_ * N_, e * N_, N_, S2, E_ * N_);
    gemm(S2, E_ * N_, xlnT_b, E_ * N_, N_, D_, E_ * N_, 1.f, 0, 0.f,
         nullptr, xe_f, D_, x1f, D_, x1b, D_);

    // transformer encoder layer (post-norm)
    gemm(x1b, D_, ipw_b + (size_t)e * 3 * D_ * D_, D_, N_, 3 * D_, D_, 1.f, 0, 0.f,
         ipb + (size_t)e * 3 * D_, nullptr, 0, nullptr, 0, qkv_b, 3 * D_);
    for (int h = 0; h < H_; h++) {
      const u16* Q = qkv_b + h * DH_;
      const u16* K = qkv_b + D_ + h * DH_;
      const u16* V = qkv_b + 2 * D_ + h * DH_;
      gemm(Q, 3 * D_, K, 3 * D_, N_, N_, DH_, 0.125f, 0, 0.f,
           nullptr, nullptr, 0, S1, N_, nullptr, 0);
      mmfs_softmax<<<N_, 256, 0, stream>>>(S1, N_, N_, 0, 0, S2, N_);
      tr(V, 3 * D_, vt_b, N_, N_, DH_);
      gemm(S2, N_, vt_b, N_, N_, DH_, N_, 1.f, 0, 0.f,
           nullptr, nullptr, 0, nullptr, 0, sa_b + h * DH_, D_);
    }
    gemm(sa_b, D_, opw_b + (size_t)e * D_ * D_, D_, N_, D_, D_, 1.f, 0, 0.f,
         opb + (size_t)e * D_, x1f, D_, t0, D_, nullptr, 0);
    mmfs_ln512<<<N_, 256, 0, stream>>>(t0, D_, n1w + (size_t)e * D_, n1b + (size_t)e * D_,
                                       x2f, D_, x2b, D_);
    gemm(x2b, D_, w1_b + (size_t)e * FF_ * D_, D_, N_, FF_, D_, 1.f, EP_RELU, 0.f,
         b1 + (size_t)e * FF_, nullptr, 0, nullptr, 0, h1_b, FF_);
    gemm(h1_b, FF_, w2_b + (size_t)e * D_ * FF_, FF_, N_, D_, FF_, 1.f, 0, 0.f,
         b2 + (size_t)e * D_, x2f, D_, t0, D_, nullptr, 0);
    mmfs_ln512<<<N_, 256, 0, stream>>>(t0, D_, n2w + (size_t)e * D_, n2b + (size_t)e * D_,
                                       x3f, D_, x3b, D_);
    tr(x3b, D_, x3Tb, N_, N_, D_);

    // hierarchical memory read + write accumulation
    for (int l = 0; l < L_; l++) {
      gemm(x3b, D_, mem_b + (size_t)l * M_ * D_, D_, N_, M_, D_, invSqrtD, EP_CLIP, 10.f,
           nullptr, nullptr, 0, S1, M_, nullptr, 0);
      mmfs_softmax<<<N_, 256, 0, stream>>>(S1, M_, M_, 0, 0, S2, M_);
      gemm(S2, M_, memT_b + (size_t)l * D_ * M_, M_, N_, D_, M_, 1.f, 0, 0.f,
           nullptr, nullptr, 0, nullptr, 0, retr_b + l * D_, L_ * D_);
      tr(S2, M_, pT_b, N_, N_, M_);
      gemm(pT_b, N_, x3Tb, N_, M_, D_, N_, 1.f, EP_CLIP, 1.f,
           nullptr, updsum + (size_t)l * M_ * D_, D_,
           updsum + (size_t)l * M_ * D_, D_, nullptr, 0);
    }

    // multi-layer aggregate + residual -> expert output
    gemm(retr_b, L_ * D_, aggw_b, L_ * D_, N_, D_, L_ * D_, 1.f, 0, 0.f,
         aggb, x3f, D_, outs + (size_t)e * ND, D_, nullptr, 0);
  }

  // ---- global gate, fusion, memory EMA ----
  mmfs_pool<<<(E_ * D_) / 256, 256, 0, stream>>>(outs, pooled);
  mmfs_gate<<<1, 128, 0, stream>>>(pooled, ggate, out_gw);
  mmfs_fuse<<<(int)(ND / 256), 256, 0, stream>>>(outs, out_gw, out_fused);
  mmfs_ema<<<4096, 256, 0, stream>>>(memsrc, updsum, out_mem, (size_t)L_ * M_ * D_);
}
